// Decoder_70248485093659
// MI455X (gfx1250) — matrix operand order repaired and checked by reading, not compiled, not run on hardware
//
#include <hip/hip_runtime.h>
#include <hip/hip_bf16.h>

typedef __attribute__((ext_vector_type(16))) _Float16 v16h;
typedef __attribute__((ext_vector_type(8)))  float    v8f;
typedef _Float16 half_t;

namespace {
constexpr int kS = 48, kP = 32, kH = 64, kT = 12;
constexpr int kN = kS * kP;
constexpr float kSlope = 0.01f;

// ---------------- LDS layout (bytes) ----------------
constexpr int OFF_FPE1 = 0;                        //   8 B-frags * 1KB (W_pe1, K=4 padded to 32, N=128)
constexpr int OFF_FPE2 = OFF_FPE1 + 8  * 1024;     //  16 B-frags (W_pe2 128x64)
constexpr int OFF_FPP1 = OFF_FPE2 + 16 * 1024;     // 128 B-frags (W_pp1 128x512)
constexpr int OFF_FPP2 = OFF_FPP1 + 128 * 1024;    //  32 B-frags (W_pp2 512x32)
constexpr int OFF_H    = OFF_FPP2 + 32 * 1024;     // h   32x64 f32
constexpr int OFF_C    = OFF_H    + 8192;          // c   32x64 f32
constexpr int OFF_DEC  = OFF_C    + 8192;          // dec_in 32x64 f32
constexpr int OFF_H16  = OFF_DEC  + 8192;          // h (f16 copy) 32x64
constexpr int OFF_LP   = OFF_H16  + 4096;          // last_pos 32x2 f32
constexpr int OFF_REL  = OFF_LP   + 256;           // rel_pos  32x2 f32
constexpr int OFF_CPOS = OFF_REL  + 256;           // curr_pos 32x2 f32
constexpr int OFF_SPD  = OFF_CPOS + 256;           // speed 32 f32
constexpr int OFF_LBL  = OFF_SPD  + 128;           // label 32 f32
constexpr int OFF_MAXV = OFF_LBL  + 128;           // pool max 32x32 f32
constexpr int OFF_BPE1 = OFF_MAXV + 4096;          // b_pe1 128 f32
constexpr int OFF_BPE2 = OFF_BPE1 + 512;           // b_pe2 64 f32
constexpr int OFF_BPP1 = OFF_BPE2 + 256;           // b_pp1 512 f32
constexpr int OFF_BPP2 = OFF_BPP1 + 2048;          // b_pp2 32 f32
constexpr int OFF_SCR  = OFF_BPP2 + 128;           // union: se1 (16KB) / m1out (8KB) / 8 * 7KB wave scratch
constexpr int WAVE_SCR = 7168;                     // t1 4KB + t2 2KB + t3 1KB per wave
constexpr int SMEM_SZ  = OFF_SCR + 8 * WAVE_SCR;   // 282496 B (< 320KB WGP LDS)
} // namespace

__device__ __forceinline__ v8f zero8() {
  v8f z;
#pragma unroll
  for (int i = 0; i < 8; ++i) z[i] = 0.0f;
  return z;
}

__device__ __forceinline__ v8f wmma_f16(v16h a, v16h b, v8f c) {
  // D(f32, 16x16) = A(f16, 16x32) * B(f16, 32x16) + C
  return __builtin_amdgcn_wmma_f32_16x16x32_f16(false, a, false, b, (short)0, c, false, false);
}

// A-fragment (16x32 f16) from a row-major f16 LDS matrix.
// lane<16: row=lane,   K halves {kb..kb+7, kb+16..kb+23} with kb=0
// lane>=16: row=lane-16, kb=8
__device__ __forceinline__ v16h load_A16(const half_t* mat, int row0, int ldm, int k0) {
  const int lane = threadIdx.x & 31;
  const int m  = lane & 15;
  const int kb = (lane & 16) ? 8 : 0;
  const half_t* r = mat + (row0 + m) * ldm + k0;
  v16h a;
#pragma unroll
  for (int i = 0; i < 8; ++i) a[i] = r[kb + i];
#pragma unroll
  for (int i = 0; i < 8; ++i) a[8 + i] = r[kb + 16 + i];
  return a;
}

// B-fragment stored pre-permuted in LDS: 512 halves, elem(lane,idx) at lane*16+idx.
__device__ __forceinline__ v16h load_Bfrag(const half_t* fragbase) {
  const half_t* p = fragbase + (threadIdx.x & 31) * 16;
  v16h b;
#pragma unroll
  for (int i = 0; i < 16; ++i) b[i] = p[i];
  return b;
}

// Write one 16x16 f32 D tile as f16 into LDS with bias (+optional LeakyReLU).
__device__ __forceinline__ void store_D_f16(half_t* out, int ldm, int col0, v8f acc,
                                            const float* bias, bool relu) {
  const int lane = threadIdx.x & 31;
  const int n  = lane & 15;
  const int mb = (lane & 16) ? 8 : 0;
  const float b = bias[n];
#pragma unroll
  for (int r = 0; r < 8; ++r) {
    float v = acc[r] + b;
    if (relu) v = (v >= 0.0f) ? v : kSlope * v;
    out[(mb + r) * ldm + col0 + n] = (half_t)v;
  }
}

// Convert a (Kreal x Ntot) f32 weight into f16 B-fragments in LDS (K zero-padded to Ktiles*32).
__device__ void stage_wfrags(half_t* dst, const float* W, int Kreal, int Ktiles, int Ntiles, int Ntot) {
  const int total = Ktiles * Ntiles * 512;
  for (int e = threadIdx.x; e < total; e += 256) {
    const int f   = e >> 9;
    const int rr  = e & 511;
    const int ln  = rr >> 4;
    const int idx = rr & 15;
    const int nt  = f % Ntiles;
    const int kt  = f / Ntiles;
    const int n   = nt * 16 + (ln & 15);
    const int k   = kt * 32 + ((ln < 16) ? 0 : 8) + ((idx < 8) ? idx : idx + 8);
    const float v = (k < Kreal) ? W[k * Ntot + n] : 0.0f;
    dst[e] = (half_t)v;
  }
}

__device__ __forceinline__ float sigf(float x) { return 1.0f / (1.0f + __expf(-x)); }
__device__ __forceinline__ float lrelu(float x) { return (x >= 0.0f) ? x : kSlope * x; }

// spatial_embedding: dec_in = lrelu([rel,spd,lbl]@W1+b1) @ W2 + b2   (scalar, tiny)
__device__ void se_embed(float* se1, const float* relp, const float* spd, const float* lbl,
                         const float* W1, const float* b1, const float* W2, const float* b2,
                         float* decin) {
  for (int e = threadIdx.x; e < 32 * 128; e += 256) {
    const int p = e >> 7, q = e & 127;
    float v = b1[q] + relp[p * 2] * W1[q] + relp[p * 2 + 1] * W1[128 + q]
            + spd[p] * W1[256 + q] + lbl[p] * W1[384 + q];
    se1[e] = lrelu(v);
  }
  __syncthreads();
  for (int e = threadIdx.x; e < 32 * 64; e += 256) {
    const int p = e >> 6, q = e & 63;
    float v = b2[q];
    for (int kk = 0; kk < 128; ++kk) v += se1[p * 128 + kk] * W2[kk * 64 + q];
    decin[p * 64 + q] = v;  // no activation (matches reference)
  }
  __syncthreads();
}

extern "C" __global__ void __launch_bounds__(256, 1)
decoder_scene_persistent(
    float* __restrict__ out,
    const float* __restrict__ last_pos, const float* __restrict__ last_pos_rel,
    const float* __restrict__ dec_h, const float* __restrict__ dec_c,
    const float* __restrict__ pps, const float* __restrict__ plb,
    const float* __restrict__ Wse1, const float* __restrict__ bse1,
    const float* __restrict__ Wse2, const float* __restrict__ bse2,
    const float* __restrict__ Wih,  const float* __restrict__ Whh,
    const float* __restrict__ bih,  const float* __restrict__ bhh,
    const float* __restrict__ Whp,  const float* __restrict__ bhp,
    const float* __restrict__ Wpe1, const float* __restrict__ bpe1,
    const float* __restrict__ Wpe2, const float* __restrict__ bpe2,
    const float* __restrict__ Wpp1, const float* __restrict__ bpp1,
    const float* __restrict__ Wpp2, const float* __restrict__ bpp2,
    const float* __restrict__ Wm1,  const float* __restrict__ bm1,
    const float* __restrict__ Wm2,  const float* __restrict__ bm2) {
  extern __shared__ char smem[];
  const int s   = blockIdx.x;     // scene
  const int tid = threadIdx.x;    // 256 threads = 8 wave32
  const int w   = tid >> 5;

  half_t* Fpe1 = (half_t*)(smem + OFF_FPE1);
  half_t* Fpe2 = (half_t*)(smem + OFF_FPE2);
  half_t* Fpp1 = (half_t*)(smem + OFF_FPP1);
  half_t* Fpp2 = (half_t*)(smem + OFF_FPP2);
  float*  hbuf = (float*)(smem + OFF_H);
  float*  cbuf = (float*)(smem + OFF_C);
  float*  decin= (float*)(smem + OFF_DEC);
  half_t* h16  = (half_t*)(smem + OFF_H16);
  float*  lp   = (float*)(smem + OFF_LP);
  float*  relp = (float*)(smem + OFF_REL);
  float*  cpos = (float*)(smem + OFF_CPOS);
  float*  spd  = (float*)(smem + OFF_SPD);
  float*  lbl  = (float*)(smem + OFF_LBL);
  float*  maxv = (float*)(smem + OFF_MAXV);
  float*  bpe1l= (float*)(smem + OFF_BPE1);
  float*  bpe2l= (float*)(smem + OFF_BPE2);
  float*  bpp1l= (float*)(smem + OFF_BPP1);
  float*  bpp2l= (float*)(smem + OFF_BPP2);
  float*  scr  = (float*)(smem + OFF_SCR);        // se1 / m1out phases
  half_t* t1   = (half_t*)(smem + OFF_SCR + w * WAVE_SCR);  // 16x128
  half_t* t2   = t1 + 16 * 128;                             // 16x64
  half_t* t3   = t2 + 16 * 64;                              // 16x32

  // ---------------- one-time staging ----------------
  stage_wfrags(Fpe1, Wpe1, 4,   1,  8, 128);
  stage_wfrags(Fpe2, Wpe2, 128, 4,  4, 64);
  stage_wfrags(Fpp1, Wpp1, 128, 4, 32, 512);
  stage_wfrags(Fpp2, Wpp2, 512, 16, 2, 32);
  for (int e = tid; e < 128; e += 256) bpe1l[e] = bpe1[e];
  for (int e = tid; e < 64;  e += 256) bpe2l[e] = bpe2[e];
  for (int e = tid; e < 512; e += 256) bpp1l[e] = bpp1[e];
  for (int e = tid; e < 32;  e += 256) bpp2l[e] = bpp2[e];
  for (int e = tid; e < 32 * 64; e += 256) {
    const int g = s * 32 * 64 + e;
    hbuf[e] = dec_h[g];
    cbuf[e] = dec_c[g];
  }
  for (int e = tid; e < 64; e += 256) {
    lp[e]   = last_pos[s * 64 + e];
    relp[e] = last_pos_rel[s * 64 + e];
  }
  if (tid < 32) {
    const int g = s * 32 + tid;
    spd[tid] = pps[g];            // pred_ped_speed[0]
    lbl[tid] = plb[g];            // pred_label[0]
  }
  __syncthreads();
  se_embed(scr, relp, spd, lbl, Wse1, bse1, Wse2, bse2, decin);  // dec_in0

  // ---------------- T-step recurrence ----------------
#pragma unroll 1
  for (int t = 0; t < kT; ++t) {
    // ---- LSTM (scalar): gates = [dec_in,h] @ [Wih;Whh] + b  -> new h,c ----
    {
      const int u = tid & 63, pg = tid >> 6;   // 4 ped-groups x 8 peds, 64 units
      float acc[8][4];
      float binit[4];
#pragma unroll
      for (int g = 0; g < 4; ++g) binit[g] = bih[g * 64 + u] + bhh[g * 64 + u];
#pragma unroll
      for (int pp = 0; pp < 8; ++pp)
#pragma unroll
        for (int g = 0; g < 4; ++g) acc[pp][g] = binit[g];
#pragma unroll 1
      for (int k = 0; k < 64; ++k) {
        const float* Wr = Wih + k * 256;
        const float w0 = Wr[u], w1 = Wr[64 + u], w2 = Wr[128 + u], w3 = Wr[192 + u];
#pragma unroll
        for (int pp = 0; pp < 8; ++pp) {
          const float z = decin[(pg * 8 + pp) * 64 + k];
          acc[pp][0] += z * w0; acc[pp][1] += z * w1;
          acc[pp][2] += z * w2; acc[pp][3] += z * w3;
        }
      }
#pragma unroll 1
      for (int k = 0; k < 64; ++k) {
        const float* Wr = Whh + k * 256;
        const float w0 = Wr[u], w1 = Wr[64 + u], w2 = Wr[128 + u], w3 = Wr[192 + u];
#pragma unroll
        for (int pp = 0; pp < 8; ++pp) {
          const float z = hbuf[(pg * 8 + pp) * 64 + k];
          acc[pp][0] += z * w0; acc[pp][1] += z * w1;
          acc[pp][2] += z * w2; acc[pp][3] += z * w3;
        }
      }
      float hn[8], cn[8];
#pragma unroll
      for (int pp = 0; pp < 8; ++pp) {
        const int p = pg * 8 + pp;
        const float c2 = sigf(acc[pp][1]) * cbuf[p * 64 + u] + sigf(acc[pp][0]) * tanhf(acc[pp][2]);
        cn[pp] = c2;
        hn[pp] = sigf(acc[pp][3]) * tanhf(c2);
      }
      __syncthreads();   // all reads of old h / dec_in done
#pragma unroll
      for (int pp = 0; pp < 8; ++pp) {
        const int p = pg * 8 + pp;
        hbuf[p * 64 + u] = hn[pp];
        cbuf[p * 64 + u] = cn[pp];
        h16[p * 64 + u]  = (half_t)hn[pp];
      }
      __syncthreads();
    }

    // ---- hidden2pos + positions + traj output; load step speed/label ----
    if (tid < 64) {
      const int p = tid >> 1, d = tid & 1;
      float v = bhp[d];
      for (int k = 0; k < 64; ++k) v += hbuf[p * 64 + k] * Whp[k * 2 + d];
      relp[p * 2 + d] = v;
      const float cp = v + lp[p * 2 + d];
      cpos[p * 2 + d] = cp;
      lp[p * 2 + d]   = cp;
      out[t * kN * 2 + (s * 32 + p) * 2 + d] = v;   // traj[t]
    }
    if (tid < 32) {
      const int g  = s * 32 + tid;
      const int ts = (t + 1 < kT) ? (t + 1) : (kT - 1);
      spd[tid] = pps[ts * kN + g];
      lbl[tid] = plb[t * kN + g];
    }
    __syncthreads();

    // ---- spatial embedding for next dec_in (scalar) ----
    se_embed(scr, relp, spd, lbl, Wse1, bse1, Wse2, bse2, decin);

    // ---- POOL: per-wave WMMA over 16-row (i, j..j+15) tiles ----
#pragma unroll 1
    for (int ii = 0; ii < 4; ++ii) {
      const int i = w + ii * 8;
      float rm0 = -1e30f, rm1 = -1e30f;
#pragma unroll 1
      for (int jh = 0; jh < 2; ++jh) {
        const int jb = jh * 16;
        // A: feat = [pos_j - pos_i, speed_j, label_j], K padded 4->32
        v16h af;
        {
          const int lane = threadIdx.x & 31;
          const int m  = lane & 15;
          const int kb = (lane & 16) ? 8 : 0;
          const int j  = jb + m;
          const float f0 = cpos[j * 2]     - cpos[i * 2];
          const float f1 = cpos[j * 2 + 1] - cpos[i * 2 + 1];
          const float f2 = spd[j], f3 = lbl[j];
#pragma unroll
          for (int idx = 0; idx < 16; ++idx) {
            const int k = (idx < 8) ? (kb + idx) : (kb + idx + 8);
            const float v = (k == 0) ? f0 : (k == 1) ? f1 : (k == 2) ? f2 : (k == 3) ? f3 : 0.0f;
            af[idx] = (half_t)v;
          }
        }
        // stage 1: t1 = lrelu(feat @ Wpe1 + b)        (16x128)
#pragma unroll
        for (int nt = 0; nt < 8; ++nt) {
          v8f acc = zero8();
          acc = wmma_f16(af, load_Bfrag(Fpe1 + nt * 512), acc);
          store_D_f16(t1, 128, nt * 16, acc, bpe1l + nt * 16, true);
        }
        // stage 2: t2 = t1 @ Wpe2 + b  (no activation) (16x64)
#pragma unroll
        for (int nt = 0; nt < 4; ++nt) {
          v8f acc = zero8();
#pragma unroll
          for (int kt = 0; kt < 4; ++kt)
            acc = wmma_f16(load_A16(t1, 0, 128, kt * 32),
                           load_Bfrag(Fpe2 + (kt * 4 + nt) * 512), acc);
          store_D_f16(t2, 64, nt * 16, acc, bpe2l + nt * 16, false);
        }
        // stage 3/4: x = [h_j | t2] (16x128); stream K of lrelu(x@Wpp1+b) into pp2 accumulators
        const v16h ah0 = load_A16(h16, jb, 64, 0);
        const v16h ah1 = load_A16(h16, jb, 64, 32);
        const v16h at0 = load_A16(t2, 0, 64, 0);
        const v16h at1 = load_A16(t2, 0, 64, 32);
        v8f acc4_0 = zero8(), acc4_1 = zero8();
#pragma unroll 1
        for (int kc = 0; kc < 16; ++kc) {
#pragma unroll
          for (int nh = 0; nh < 2; ++nh) {
            const int nt = kc * 2 + nh;
            v8f acc = zero8();
            acc = wmma_f16(ah0, load_Bfrag(Fpp1 + (0 * 32 + nt) * 512), acc);
            acc = wmma_f16(ah1, load_Bfrag(Fpp1 + (1 * 32 + nt) * 512), acc);
            acc = wmma_f16(at0, load_Bfrag(Fpp1 + (2 * 32 + nt) * 512), acc);
            acc = wmma_f16(at1, load_Bfrag(Fpp1 + (3 * 32 + nt) * 512), acc);
            store_D_f16(t3, 32, nh * 16, acc, bpp1l + nt * 16, true);
          }
          const v16h a3 = load_A16(t3, 0, 32, 0);
          acc4_0 = wmma_f16(a3, load_Bfrag(Fpp2 + (kc * 2 + 0) * 512), acc4_0);
          acc4_1 = wmma_f16(a3, load_Bfrag(Fpp2 + (kc * 2 + 1) * 512), acc4_1);
        }
        // bias + lrelu + max over the 16 rows (all share i)
        const int lane = threadIdx.x & 31;
        const int n = lane & 15;
        {
          const float b = bpp2l[n];
          float mx = -1e30f;
#pragma unroll
          for (int r = 0; r < 8; ++r) mx = fmaxf(mx, lrelu(acc4_0[r] + b));
          mx = fmaxf(mx, __shfl_xor(mx, 16));
          rm0 = fmaxf(rm0, mx);
        }
        {
          const float b = bpp2l[16 + n];
          float mx = -1e30f;
#pragma unroll
          for (int r = 0; r < 8; ++r) mx = fmaxf(mx, lrelu(acc4_1[r] + b));
          mx = fmaxf(mx, __shfl_xor(mx, 16));
          rm1 = fmaxf(rm1, mx);
        }
      }
      const int lane = threadIdx.x & 31;
      if (lane < 16) {
        maxv[i * 32 + lane]      = rm0;
        maxv[i * 32 + 16 + lane] = rm1;
      }
    }
    __syncthreads();

    // ---- decoder MLP: h = lrelu(lrelu([h|pool] @ Wm1 + b) @ Wm2 + b) ----
    for (int e = tid; e < 32 * 64; e += 256) {
      const int p = e >> 6, q = e & 63;
      float v = bm1[q];
      for (int k = 0; k < 64; ++k) v += hbuf[p * 64 + k] * Wm1[k * 64 + q];
      for (int k = 0; k < 32; ++k) v += maxv[p * 32 + k] * Wm1[(64 + k) * 64 + q];
      scr[e] = lrelu(v);
    }
    __syncthreads();
    for (int e = tid; e < 32 * 64; e += 256) {
      const int p = e >> 6, q = e & 63;
      float v = bm2[q];
      for (int k = 0; k < 64; ++k) v += scr[p * 64 + k] * Wm2[k * 64 + q];
      hbuf[e] = lrelu(v);
    }
    __syncthreads();
  }

  // ---- final hidden state output ----
  for (int e = tid; e < 32 * 64; e += 256)
    out[kT * kN * 2 + s * 32 * 64 + e] = hbuf[e];
}

extern "C" void kernel_launch(void* const* d_in, const int* in_sizes, int n_in,
                              void* d_out, int out_size, void* d_ws, size_t ws_size,
                              hipStream_t stream) {
  (void)in_sizes; (void)n_in; (void)out_size; (void)d_ws; (void)ws_size;
  const float* last_pos     = (const float*)d_in[0];
  const float* last_pos_rel = (const float*)d_in[1];
  const float* dec_h        = (const float*)d_in[2];
  const float* dec_c        = (const float*)d_in[3];
  const float* pps          = (const float*)d_in[4];
  const float* plb          = (const float*)d_in[5];
  // d_in[6..8]: seq_start_end / train_or_test / speed_to_add — uniform groups, unused
  const float* Wse1 = (const float*)d_in[9],  *bse1 = (const float*)d_in[10];
  const float* Wse2 = (const float*)d_in[11], *bse2 = (const float*)d_in[12];
  const float* Wih  = (const float*)d_in[13], *Whh  = (const float*)d_in[14];
  const float* bih  = (const float*)d_in[15], *bhh  = (const float*)d_in[16];
  const float* Whp  = (const float*)d_in[17], *bhp  = (const float*)d_in[18];
  const float* Wpe1 = (const float*)d_in[19], *bpe1 = (const float*)d_in[20];
  const float* Wpe2 = (const float*)d_in[21], *bpe2 = (const float*)d_in[22];
  const float* Wpp1 = (const float*)d_in[23], *bpp1 = (const float*)d_in[24];
  const float* Wpp2 = (const float*)d_in[25], *bpp2 = (const float*)d_in[26];
  const float* Wm1  = (const float*)d_in[27], *bm1  = (const float*)d_in[28];
  const float* Wm2  = (const float*)d_in[29], *bm2  = (const float*)d_in[30];

  (void)hipFuncSetAttribute((const void*)decoder_scene_persistent,
                            hipFuncAttributeMaxDynamicSharedMemorySize, SMEM_SZ);
  decoder_scene_persistent<<<kS, 256, SMEM_SZ, stream>>>(
      (float*)d_out, last_pos, last_pos_rel, dec_h, dec_c, pps, plb,
      Wse1, bse1, Wse2, bse2, Wih, Whh, bih, bhh, Whp, bhp,
      Wpe1, bpe1, Wpe2, bpe2, Wpp1, bpp1, Wpp2, bpp2,
      Wm1, bm1, Wm2, bm2);
}
